// SpatialLocalAttention_26774826123576
// MI455X (gfx1250) — matrix operand order repaired and checked by reading, not compiled, not run on hardware
//
#include <hip/hip_runtime.h>
#include <hip/hip_bf16.h>
#include <math.h>

// ---------------------------------------------------------------------------
// Problem constants (B=2, L=2048, D=512, K=32, G=32, H=8, DH=64, INNER=512)
// ---------------------------------------------------------------------------
#define BB     2
#define LL     2048
#define DD     512
#define KK     32
#define GG     32
#define HH     8
#define DHH    64
#define INNER  512
#define NSPAT  (BB * LL)          // 4096 spatial rows
#define NGLOB  (BB * GG)          // 64 global rows
#define NALL   (NSPAT + NGLOB)    // 4160
#define CTX    (1 + KK + GG)      // 65 context slots

typedef __attribute__((ext_vector_type(16))) __bf16 v16bf;
typedef __attribute__((ext_vector_type(8)))  float  v8f;

// ---------------------------------------------------------------------------
// One-off weight pack: Wt[n][k] = bf16(W[k][n]).  LDS 32x32 tile transpose,
// coalesced on both sides.  After this, a WMMA B-fragment (16 K-rows of one
// column) is 16 CONTIGUOUS bf16 = two b128 loads.
// ---------------------------------------------------------------------------
__global__ __launch_bounds__(256)
void pack_w_t(const float* __restrict__ W, __bf16* __restrict__ Wt)
{
    __shared__ float tile[32][33];
    const int bx = blockIdx.x & 15;    // k-tile
    const int by = blockIdx.x >> 4;    // n-tile
    const int tx = threadIdx.x & 31;
    const int ty = threadIdx.x >> 5;   // 0..7
    #pragma unroll
    for (int r = ty; r < 32; r += 8)
        tile[r][tx] = W[(size_t)(bx * 32 + r) * INNER + by * 32 + tx];
    __syncthreads();
    #pragma unroll
    for (int r = ty; r < 32; r += 8)
        Wt[(size_t)(by * 32 + r) * DD + bx * 32 + tx] = (__bf16)tile[tx][r];
}

// ---------------------------------------------------------------------------
// WMMA GEMM:  out[M x 512] = X[M x 512] @ W[512 x 512] (+ bias),
// with W pre-transposed/packed to bf16 (Wt[n][k]).
// X rows 0..4095 come from X0 (spatial), rows 4096.. from X1 (globals).
//
// Block (256 thr, 8 waves) owns one 16-row x 512-col output stripe.
// Per 32-wide K-chunk: block stages the 16x32 A panel into LDS as bf16
// (double buffered, one barrier per chunk); each wave reads one A fragment
// (2 x ds_load_b128) and runs FOUR v_wmma_f32_16x16x32_bf16 against 4 B
// fragments (2 x b128 loads each from Wt) -> 4 accumulators.
// Fragment K order: contiguous, k = 16*half + s, same in A and B.
// ---------------------------------------------------------------------------
__global__ __launch_bounds__(256)
void wmma_gemm_bf16(const float* __restrict__ X0, const float* __restrict__ X1,
                    const __bf16* __restrict__ Wt,
                    const float* __restrict__ bias, float* __restrict__ out)
{
    __shared__ unsigned AsU[2][16][16];   // [buf][row][k-pair] : 16x32 bf16

    const int tid  = threadIdx.x;
    const int lane = tid & 31;
    const int wave = tid >> 5;            // 0..7 -> 64-col slice
    const int m0   = blockIdx.x * 16;
    const int n0   = wave * 64;
    const int half = lane >> 4;
    const int mr   = lane & 15;           // A row within tile
    const int nc   = lane & 15;           // B/C/D column within 16-wide subtile

    // --- per-thread A staging assignment: row = tid/16, k-pair = tid%16 ---
    const int arow = tid >> 4;
    const int acol = tid & 15;
    const int grow = m0 + arow;
    const float* xr = ((grow < NSPAT) ? (X0 + (size_t)grow * DD)
                                      : (X1 + (size_t)(grow - NSPAT) * DD))
                      + acol * 2;

    v8f acc[4] = {};

    for (int kk = 0; kk < DD; kk += 32) {
        const int p  = (kk >> 5) & 1;
        const int pk = (kk + 32 < DD) ? (kk + 32) : kk;   // clamped, branch-free

        // ---- stage A panel chunk into LDS (fp32 -> packed bf16x2) ----
        {
            const float2 f = *(const float2*)(xr + kk);
            union { __bf16 h[2]; unsigned u; } cv;
            cv.h[0] = (__bf16)f.x;
            cv.h[1] = (__bf16)f.y;
            AsU[p][arow][acol] = cv.u;
        }
        __builtin_prefetch(xr + pk, 0, 1);    // next A chunk (unconditional)
        __syncthreads();

        // ---- A fragment from LDS: rows mr, K = half*16..+15
        v16bf a;
        {
            union { uint4 v; __bf16 h[8]; } f0, f1;
            f0.v = *(const uint4*)&AsU[p][mr][half * 8];
            f1.v = *(const uint4*)&AsU[p][mr][half * 8 + 4];
            #pragma unroll
            for (int t = 0; t < 8; ++t) { a[t] = f0.h[t]; a[8 + t] = f1.h[t]; }
        }

        // ---- 4 B fragments (contiguous bf16 from Wt) + 4 WMMAs ----
        #pragma unroll
        for (int j = 0; j < 4; ++j) {
            const __bf16* wp = Wt + (size_t)(n0 + j * 16 + nc) * DD
                                  + kk + half * 16;
            __builtin_prefetch(wp + (pk - kk), 0, 1);   // next chunk, same row
            union { uint4 v; __bf16 h[8]; } b0, b1;
            b0.v = *(const uint4*)wp;          // K = kb .. kb+7
            b1.v = *(const uint4*)(wp + 8);    // K = kb+8 .. kb+15
            v16bf bf;
            #pragma unroll
            for (int t = 0; t < 8; ++t) { bf[t] = b0.h[t]; bf[8 + t] = b1.h[t]; }
            acc[j] = __builtin_amdgcn_wmma_f32_16x16x32_bf16(
                         false, a, false, bf, (short)0, acc[j], false, false);
        }
    }

    // ---- epilogue: bias hoisted out of the store loop ----
    float bv[4] = {0.0f, 0.0f, 0.0f, 0.0f};
    if (bias) {
        #pragma unroll
        for (int j = 0; j < 4; ++j) bv[j] = bias[n0 + j * 16 + nc];
    }
    #pragma unroll
    for (int j = 0; j < 4; ++j) {
        const int c = n0 + j * 16 + nc;
        #pragma unroll
        for (int v = 0; v < 8; ++v) {
            const int r = m0 + v + 8 * half;
            out[(size_t)r * INNER + c] = acc[j][v] + bv[j];
        }
    }
}

// ---------------------------------------------------------------------------
// Attention: one wave32 per (b, l, h).
// Phase 1: lanes own context slots c = lane, lane+32, (lane==0: 64);
//          dot(Q,K) over DH=64 + distance / global bias.
// Cross-lane max & sum via __shfl_xor (wave32).
// Phase 2: lanes own dh = lane, lane+32; loop c, broadcast p[c] via __shfl.
// ---------------------------------------------------------------------------
__global__ __launch_bounds__(256)
void attn_kernel(const float* __restrict__ Q, const float* __restrict__ Kp,
                 const float* __restrict__ Vp, const int* __restrict__ topk,
                 const float* __restrict__ dist,
                 const float* __restrict__ log_sigma,
                 const float* __restrict__ global_bias,
                 float* __restrict__ attnout)
{
    const int lane = threadIdx.x & 31;
    const int wave = threadIdx.x >> 5;
    const int idx  = blockIdx.x * 8 + wave;     // 0 .. B*L*H-1
    const int h    = idx & (HH - 1);
    const int l    = (idx >> 3) & (LL - 1);
    const int b    = idx >> 14;                 // / (H*L)
    const int bl   = b * LL + l;

    const float scale  = 0.125f;                // DH^-0.5
    const float sig    = __expf(log_sigma[h]);
    const float inv2s2 = 1.0f / (2.0f * sig * sig);
    const float gb     = global_bias[0];
    const float* qrow  = Q + (size_t)bl * INNER + h * DHH;

    float logit[3];
    #pragma unroll
    for (int s = 0; s < 3; ++s) {
        const int c = lane + 32 * s;
        float lg = -__builtin_inff();
        if (c < CTX) {
            int   krow;
            float biasv;
            if (c == 0) {
                krow = bl; biasv = 0.0f;
            } else if (c <= KK) {
                const int nb = topk[(size_t)bl * KK + (c - 1)];
                krow = b * LL + nb;
                const float dd = dist[(size_t)bl * KK + (c - 1)];
                biasv = -dd * dd * inv2s2;
            } else {
                krow = NSPAT + b * GG + (c - 1 - KK);
                biasv = gb;
            }
            const float* kr = Kp + (size_t)krow * INNER + h * DHH;
            float acc = 0.0f;
            #pragma unroll
            for (int t = 0; t < 16; ++t) {
                const float4 qv = ((const float4*)qrow)[t];
                const float4 kv = ((const float4*)kr)[t];
                acc += qv.x * kv.x + qv.y * kv.y + qv.z * kv.z + qv.w * kv.w;
            }
            lg = acc * scale + biasv;
        }
        logit[s] = lg;
    }

    // wave32 max-reduction
    float mx = fmaxf(fmaxf(logit[0], logit[1]), logit[2]);
    #pragma unroll
    for (int off = 16; off > 0; off >>= 1)
        mx = fmaxf(mx, __shfl_xor(mx, off, 32));

    float e[3];
    float ssum = 0.0f;
    #pragma unroll
    for (int s = 0; s < 3; ++s) {
        e[s] = (logit[s] <= -__builtin_inff()) ? 0.0f : __expf(logit[s] - mx);
        ssum += e[s];
    }
    #pragma unroll
    for (int off = 16; off > 0; off >>= 1)
        ssum += __shfl_xor(ssum, off, 32);
    const float inv = 1.0f / ssum;

    // Phase 2: weighted V sum; lanes own dh = lane, lane+32
    float acc0 = 0.0f, acc1 = 0.0f;
    for (int c = 0; c < CTX; ++c) {
        const int s = c >> 5;                       // uniform
        float pv = (s == 0) ? e[0] : ((s == 1) ? e[1] : e[2]);
        pv = __shfl(pv, c & 31, 32) * inv;
        int vrow;
        if (c == 0)        vrow = bl;
        else if (c <= KK)  vrow = b * LL + topk[(size_t)bl * KK + (c - 1)];
        else               vrow = NSPAT + b * GG + (c - 1 - KK);
        const float* vr = Vp + (size_t)vrow * INNER + h * DHH;
        acc0 += pv * vr[lane];
        acc1 += pv * vr[lane + 32];
    }
    float* orow = attnout + (size_t)bl * INNER + h * DHH;
    orow[lane]      = acc0;
    orow[lane + 32] = acc1;
}

// ---------------------------------------------------------------------------
// Launcher.  d_in order: spatial, topk_indices, distances, global_latents,
//            Wq, Wk, Wv, Wo, bo, log_sigma, global_bias
// Workspace (floats): Q[4096*512] | K[4160*512] | V[4160*512] | A[4096*512]
// then (bf16): WqT | WkT | WvT | WoT, each 512*512
// ---------------------------------------------------------------------------
extern "C" void kernel_launch(void* const* d_in, const int* in_sizes, int n_in,
                              void* d_out, int out_size, void* d_ws, size_t ws_size,
                              hipStream_t stream)
{
    (void)in_sizes; (void)n_in; (void)out_size; (void)ws_size;

    const float* spatial = (const float*)d_in[0];
    const int*   topk    = (const int*)  d_in[1];
    const float* dist    = (const float*)d_in[2];
    const float* glob    = (const float*)d_in[3];
    const float* Wq      = (const float*)d_in[4];
    const float* Wk      = (const float*)d_in[5];
    const float* Wv      = (const float*)d_in[6];
    const float* Wo      = (const float*)d_in[7];
    const float* bo      = (const float*)d_in[8];
    const float* lsig    = (const float*)d_in[9];
    const float* gbias   = (const float*)d_in[10];
    float*       out     = (float*)d_out;

    float* ws = (float*)d_ws;
    float* Qb = ws;                                   // 4096*512
    float* Kb = Qb + (size_t)NSPAT * INNER;           // 4160*512
    float* Vb = Kb + (size_t)NALL  * INNER;           // 4160*512
    float* Ab = Vb + (size_t)NALL  * INNER;           // 4096*512
    __bf16* WqT = (__bf16*)(Ab + (size_t)NSPAT * INNER);
    __bf16* WkT = WqT + (size_t)INNER * DD;
    __bf16* WvT = WkT + (size_t)INNER * DD;
    __bf16* WoT = WvT + (size_t)INNER * DD;

    // one-off bf16 transpose-pack of the four weight matrices
    pack_w_t<<<256, 256, 0, stream>>>(Wq, WqT);
    pack_w_t<<<256, 256, 0, stream>>>(Wk, WkT);
    pack_w_t<<<256, 256, 0, stream>>>(Wv, WvT);
    pack_w_t<<<256, 256, 0, stream>>>(Wo, WoT);

    // one block per 16-row output stripe
    wmma_gemm_bf16<<<NSPAT / 16, 256, 0, stream>>>(spatial, nullptr, WqT, nullptr, Qb);
    wmma_gemm_bf16<<<NALL  / 16, 256, 0, stream>>>(spatial, glob,    WkT, nullptr, Kb);
    wmma_gemm_bf16<<<NALL  / 16, 256, 0, stream>>>(spatial, glob,    WvT, nullptr, Vb);

    const int attnBlocks = (BB * LL * HH) / 8;   // 4096
    attn_kernel<<<attnBlocks, 256, 0, stream>>>(Qb, Kb, Vb, topk, dist, lsig, gbias, Ab);

    wmma_gemm_bf16<<<NSPAT / 16, 256, 0, stream>>>(Ab, nullptr, WoT, bo, out);
}
